// Attention_41343355191713
// MI455X (gfx1250) — compile-verified, matrix operand order checked by reading, not run on hardware
//
#include <hip/hip_runtime.h>
#include <hip/hip_bf16.h>

// ---------------------------------------------------------------------------
// CDNA5 (gfx1250) graph-attention block. WMMA bf16 GEMMs, f32 accumulation.
// Weights pre-packed into per-lane fragment order -> B128 vector loads.
// ---------------------------------------------------------------------------

typedef __attribute__((ext_vector_type(16))) __bf16 v16bf;
typedef __attribute__((ext_vector_type(8)))  float  v8f;

#define DIM       128
#define HEADS     4
#define HEAD_DIM  32
#define QKV_COLS  384
#define LN_EPS    1e-6f
#define ATT_SCALE 0.17677669529663687f   // 1/sqrt(32)

// ---- bf16 helpers on raw ushort storage (avoid scalar __bf16 arithmetic) ---
__device__ __forceinline__ float bf_lo(unsigned int u) { return __uint_as_float(u << 16); }
__device__ __forceinline__ float bf_hi(unsigned int u) { return __uint_as_float(u & 0xFFFF0000u); }
__device__ __forceinline__ unsigned short f2bf(float f) {
    unsigned int u = __float_as_uint(f);
    u += 0x7FFFu + ((u >> 16) & 1u);          // round-to-nearest-even
    return (unsigned short)(u >> 16);
}

// Fragment element -> K index mapping (16-bit A/B operand, 16x32, wave32).
// Lane l (l<16): K = {0..7, 16..23}; lane l+16: K = {8..15, 24..31}.
__device__ __forceinline__ int wmma_k_of(int v, int hlf) {
    int p = v >> 1, e = v & 1;
    int base = (p < 4) ? (p * 2 + e) : (16 + (p - 4) * 2 + e);
    return base + hlf * 8;
}

__device__ __forceinline__ void atomicMaxF(float* addr, float val) {
    int old = __float_as_int(*addr);
    while (__int_as_float(old) < val) {
        int prev = atomicCAS((int*)addr, old, __float_as_int(val));
        if (prev == old) break;
        old = prev;
    }
}

// ---------------------------- utility kernels ------------------------------
__global__ void k_cvt_bf16(const float* __restrict__ src, unsigned short* __restrict__ dst, int n) {
    int t = blockIdx.x * blockDim.x + threadIdx.x;
    if (t < n) dst[t] = f2bf(src[t]);
}

__global__ void k_fill(float* __restrict__ p, float v, int n) {
    int t = blockIdx.x * blockDim.x + threadIdx.x;
    if (t < n) p[t] = v;
}

// Pack W[K=128, NC] (f32, row-major) into WMMA B-fragment order (bf16):
// wpk[(((kb*ntiles + tn)*32 + lane)*16 + v] = bf16(W[kb*32 + k_of(v, lane>>4)][tn*16 + (lane&15)])
// One thread per packed element; total = DIM * NC.
__global__ void k_pack_w(const float* __restrict__ W, unsigned short* __restrict__ wpk, int NC) {
    int t = blockIdx.x * blockDim.x + threadIdx.x;
    if (t >= DIM * NC) return;
    int v    = t & 15;
    int lane = (t >> 4) & 31;
    int rest = t >> 9;                 // kb*ntiles + tn
    int ntiles = NC >> 4;
    int tn = rest % ntiles;
    int kb = rest / ntiles;
    int k  = kb * 32 + wmma_k_of(v, lane >> 4);
    int n  = tn * 16 + (lane & 15);
    wpk[t] = f2bf(W[(size_t)k * NC + n]);
}

// ------------------------------ QKV GEMM -----------------------------------
// qkv[N,384] = h[N,128] @ W[128,384] + b ; output stored bf16.
// Block = 128 threads (4 waves). Wave w: row tile = blockIdx.x*4+w,
// col strip = 4 consecutive tiles at blockIdx.y*4. 16 WMMAs per wave.
__global__ void k_gemm_qkv(const unsigned short* __restrict__ hb,
                           const unsigned short* __restrict__ wpk,
                           const float* __restrict__ bias,
                           unsigned short* __restrict__ out) {
    const int lane   = threadIdx.x & 31;
    const int wave   = threadIdx.x >> 5;
    const int tile_m = blockIdx.x * 4 + wave;
    const int tn0    = blockIdx.y * 4;           // first of 4 col tiles
    const int hlf    = lane >> 4;
    const int l15    = lane & 15;
    const int m      = tile_m * 16 + l15;
    const int ntiles = QKV_COLS / 16;            // 24

    v8f acc0 = {}, acc1 = {}, acc2 = {}, acc3 = {};
    #pragma unroll
    for (int kb = 0; kb < 4; ++kb) {
        const int k0 = kb * 32;
        union { uint4 u[2]; v16bf v; } a;
        a.u[0] = *(const uint4*)(hb + (size_t)m * DIM + k0 + hlf * 8);
        a.u[1] = *(const uint4*)(hb + (size_t)m * DIM + k0 + 16 + hlf * 8);
        union { uint4 u[2]; v16bf v; } b0, b1, b2, b3;
        const uint4* bp = (const uint4*)(wpk + (((size_t)kb * ntiles + tn0) * 32 + lane) * 16);
        b0.u[0] = bp[0];  b0.u[1] = bp[1];       // tile tn0+0 (each tile = 32 lanes * 32B)
        b1.u[0] = bp[64]; b1.u[1] = bp[65];      // tile tn0+1: +32*32B = +64 uint4
        b2.u[0] = bp[128]; b2.u[1] = bp[129];
        b3.u[0] = bp[192]; b3.u[1] = bp[193];
        acc0 = __builtin_amdgcn_wmma_f32_16x16x32_bf16(false, a.v, false, b0.v, (short)0, acc0, false, false);
        acc1 = __builtin_amdgcn_wmma_f32_16x16x32_bf16(false, a.v, false, b1.v, (short)0, acc1, false, false);
        acc2 = __builtin_amdgcn_wmma_f32_16x16x32_bf16(false, a.v, false, b2.v, (short)0, acc2, false, false);
        acc3 = __builtin_amdgcn_wmma_f32_16x16x32_bf16(false, a.v, false, b3.v, (short)0, acc3, false, false);
    }
    #pragma unroll
    for (int nt = 0; nt < 4; ++nt) {
        v8f acc = (nt == 0) ? acc0 : (nt == 1) ? acc1 : (nt == 2) ? acc2 : acc3;
        const int n = (tn0 + nt) * 16 + l15;
        const float bn = bias[n];
        #pragma unroll
        for (int v2 = 0; v2 < 8; ++v2) {
            int mg = tile_m * 16 + v2 + hlf * 8;
            out[(size_t)mg * QKV_COLS + n] = f2bf(acc[v2] + bn);
        }
    }
}

// ------------------------------ MLP GEMM -----------------------------------
// h2[N,128] = h1 + silu(h1 @ W_mlp + b) ; fused epilogue.
// Same tiling; blockIdx.y==0 covers all 8 col tiles in 2 strips? No:
// grid.y = 2 strips of 4 tiles.
__global__ void k_gemm_mlp(const unsigned short* __restrict__ h1b,
                           const unsigned short* __restrict__ wpk,
                           const float* __restrict__ bias,
                           const float* __restrict__ h1,
                           float* __restrict__ h2) {
    const int lane   = threadIdx.x & 31;
    const int wave   = threadIdx.x >> 5;
    const int tile_m = blockIdx.x * 4 + wave;
    const int tn0    = blockIdx.y * 4;
    const int hlf    = lane >> 4;
    const int l15    = lane & 15;
    const int m      = tile_m * 16 + l15;
    const int ntiles = DIM / 16;                 // 8

    v8f acc0 = {}, acc1 = {}, acc2 = {}, acc3 = {};
    #pragma unroll
    for (int kb = 0; kb < 4; ++kb) {
        const int k0 = kb * 32;
        union { uint4 u[2]; v16bf v; } a;
        a.u[0] = *(const uint4*)(h1b + (size_t)m * DIM + k0 + hlf * 8);
        a.u[1] = *(const uint4*)(h1b + (size_t)m * DIM + k0 + 16 + hlf * 8);
        union { uint4 u[2]; v16bf v; } b0, b1, b2, b3;
        const uint4* bp = (const uint4*)(wpk + (((size_t)kb * ntiles + tn0) * 32 + lane) * 16);
        b0.u[0] = bp[0];   b0.u[1] = bp[1];
        b1.u[0] = bp[64];  b1.u[1] = bp[65];
        b2.u[0] = bp[128]; b2.u[1] = bp[129];
        b3.u[0] = bp[192]; b3.u[1] = bp[193];
        acc0 = __builtin_amdgcn_wmma_f32_16x16x32_bf16(false, a.v, false, b0.v, (short)0, acc0, false, false);
        acc1 = __builtin_amdgcn_wmma_f32_16x16x32_bf16(false, a.v, false, b1.v, (short)0, acc1, false, false);
        acc2 = __builtin_amdgcn_wmma_f32_16x16x32_bf16(false, a.v, false, b2.v, (short)0, acc2, false, false);
        acc3 = __builtin_amdgcn_wmma_f32_16x16x32_bf16(false, a.v, false, b3.v, (short)0, acc3, false, false);
    }
    #pragma unroll
    for (int nt = 0; nt < 4; ++nt) {
        v8f acc = (nt == 0) ? acc0 : (nt == 1) ? acc1 : (nt == 2) ? acc2 : acc3;
        const int n = (tn0 + nt) * 16 + l15;
        const float bn = bias[n];
        #pragma unroll
        for (int v2 = 0; v2 < 8; ++v2) {
            int mg = tile_m * 16 + v2 + hlf * 8;
            float z = acc[v2] + bn;
            float s = z / (1.0f + __expf(-z));   // silu
            h2[(size_t)mg * DIM + n] = h1[(size_t)mg * DIM + n] + s;
        }
    }
}

// --------------------------- edge pass 1: scores ---------------------------
__global__ void k_edge_score(const unsigned short* __restrict__ qkv,
                             const int* __restrict__ ei, const int* __restrict__ ej,
                             float* __restrict__ A, float* __restrict__ segmax, int total) {
    int t = blockIdx.x * blockDim.x + threadIdx.x;
    if (t >= total) return;
    int e = t >> 2, h = t & 3;
    int i = ei[e], j = ej[e];
    const uint4* q4 = (const uint4*)(qkv + (size_t)i * QKV_COLS + h * HEAD_DIM);
    const uint4* k4 = (const uint4*)(qkv + (size_t)j * QKV_COLS + DIM + h * HEAD_DIM);
    float s = 0.0f;
    #pragma unroll
    for (int c = 0; c < 4; ++c) {
        uint4 qa = q4[c], kb = k4[c];
        s += bf_lo(qa.x) * bf_lo(kb.x) + bf_hi(qa.x) * bf_hi(kb.x);
        s += bf_lo(qa.y) * bf_lo(kb.y) + bf_hi(qa.y) * bf_hi(kb.y);
        s += bf_lo(qa.z) * bf_lo(kb.z) + bf_hi(qa.z) * bf_hi(kb.z);
        s += bf_lo(qa.w) * bf_lo(kb.w) + bf_hi(qa.w) * bf_hi(kb.w);
    }
    s *= ATT_SCALE;
    A[t] = s;
    atomicMaxF(segmax + (size_t)j * HEADS + h, s);
}

// --------------------------- edge pass 2: exp + sum ------------------------
__global__ void k_edge_exp(const int* __restrict__ ej, float* __restrict__ A,
                           const float* __restrict__ segmax, float* __restrict__ denom, int total) {
    int t = blockIdx.x * blockDim.x + threadIdx.x;
    if (t >= total) return;
    int e = t >> 2, h = t & 3;
    int j = ej[e];
    float ex = __expf(A[t] - segmax[(size_t)j * HEADS + h]);
    A[t] = ex;
    atomicAdd(denom + (size_t)j * HEADS + h, ex);
}

// ---------------------- edge pass 3: normalize + scatter -------------------
__global__ void k_edge_scatter(const unsigned short* __restrict__ qkv,
                               const int* __restrict__ ei, const int* __restrict__ ej,
                               const float* __restrict__ A, const float* __restrict__ denom,
                               float* __restrict__ attn, int total) {
    int t = blockIdx.x * blockDim.x + threadIdx.x;
    if (t >= total) return;
    int e = t >> 2, h = t & 3;
    int i = ei[e], j = ej[e];
    float w = A[t] / denom[(size_t)j * HEADS + h];
    const uint4* v4 = (const uint4*)(qkv + (size_t)j * QKV_COLS + 2 * DIM + h * HEAD_DIM);
    float* dst = attn + (size_t)i * DIM + h * HEAD_DIM;
    #pragma unroll
    for (int c = 0; c < 4; ++c) {
        uint4 va = v4[c];
        atomicAdd(dst + c * 8 + 0, w * bf_lo(va.x));
        atomicAdd(dst + c * 8 + 1, w * bf_hi(va.x));
        atomicAdd(dst + c * 8 + 2, w * bf_lo(va.y));
        atomicAdd(dst + c * 8 + 3, w * bf_hi(va.y));
        atomicAdd(dst + c * 8 + 4, w * bf_lo(va.z));
        atomicAdd(dst + c * 8 + 5, w * bf_hi(va.z));
        atomicAdd(dst + c * 8 + 6, w * bf_lo(va.w));
        atomicAdd(dst + c * 8 + 7, w * bf_hi(va.w));
    }
}

// --------------------- residual + LayerNorm (1 warp/row) -------------------
__global__ void k_add_ln(const float* __restrict__ a, const float* __restrict__ b,
                         const float* __restrict__ scale, const float* __restrict__ bias,
                         float* __restrict__ out, unsigned short* __restrict__ out_bf) {
    int lane = threadIdx.x & 31;
    int warp = threadIdx.x >> 5;
    int row  = blockIdx.x * 4 + warp;

    float4 x = ((const float4*)(a + (size_t)row * DIM))[lane];
    if (b) {
        float4 y = ((const float4*)(b + (size_t)row * DIM))[lane];
        x.x += y.x; x.y += y.y; x.z += y.z; x.w += y.w;
    }
    float s = x.x + x.y + x.z + x.w;
    #pragma unroll
    for (int m = 16; m; m >>= 1) s += __shfl_xor(s, m, 32);
    float mu = s * (1.0f / DIM);
    float d0 = x.x - mu, d1 = x.y - mu, d2 = x.z - mu, d3 = x.w - mu;
    float v = d0 * d0 + d1 * d1 + d2 * d2 + d3 * d3;
    #pragma unroll
    for (int m = 16; m; m >>= 1) v += __shfl_xor(v, m, 32);
    float rs = rsqrtf(v * (1.0f / DIM) + LN_EPS);
    float4 sc = ((const float4*)scale)[lane];
    float4 bi = ((const float4*)bias)[lane];
    float4 y;
    y.x = d0 * rs * sc.x + bi.x;
    y.y = d1 * rs * sc.y + bi.y;
    y.z = d2 * rs * sc.z + bi.z;
    y.w = d3 * rs * sc.w + bi.w;
    ((float4*)(out + (size_t)row * DIM))[lane] = y;
    if (out_bf) {
        unsigned short* ob = out_bf + (size_t)row * DIM + lane * 4;
        ob[0] = f2bf(y.x); ob[1] = f2bf(y.y); ob[2] = f2bf(y.z); ob[3] = f2bf(y.w);
    }
}

// ---------------------------------------------------------------------------
extern "C" void kernel_launch(void* const* d_in, const int* in_sizes, int n_in,
                              void* d_out, int out_size, void* d_ws, size_t ws_size,
                              hipStream_t stream) {
    const float* h_one = (const float*)d_in[0];
    const float* W_qkv = (const float*)d_in[1];
    const float* b_qkv = (const float*)d_in[2];
    const float* ln1_s = (const float*)d_in[3];
    const float* ln1_b = (const float*)d_in[4];
    const float* W_mlp = (const float*)d_in[5];
    const float* b_mlp = (const float*)d_in[6];
    const float* ln2_s = (const float*)d_in[7];
    const float* ln2_b = (const float*)d_in[8];
    const int*   e_i   = (const int*)d_in[9];
    const int*   e_j   = (const int*)d_in[10];
    float* out = (float*)d_out;

    const int N = in_sizes[0] / DIM;     // 65536
    const int E = in_sizes[9];           // 2097152
    const int EH = E * HEADS;

    // ---- workspace carve-up (all chunks 256B-aligned) ----
    char* ws = (char*)d_ws;
    size_t off = 0;
    auto take = [&](size_t bytes) { size_t r = off; off += (bytes + 255) & ~(size_t)255; return r; };
    unsigned short* hb    = (unsigned short*)(ws + take((size_t)N * DIM * 2));        // h_one bf16
    unsigned short* wqkvp = (unsigned short*)(ws + take((size_t)DIM * QKV_COLS * 2)); // W_qkv packed bf16
    unsigned short* wmlpp = (unsigned short*)(ws + take((size_t)DIM * DIM * 2));      // W_mlp packed bf16
    unsigned short* qkvb  = (unsigned short*)(ws + take((size_t)N * QKV_COLS * 2));   // QKV bf16
    float* A      = (float*)(ws + take((size_t)EH * 4));                              // edge scores
    float* segmax = (float*)(ws + take((size_t)N * HEADS * 4));
    float* denom  = (float*)(ws + take((size_t)N * HEADS * 4));
    float* attn   = (float*)(ws + take((size_t)N * DIM * 4));
    float* h1     = (float*)(ws + take((size_t)N * DIM * 4));
    unsigned short* h1b = hb;        // overlay: hb dead after QKV gemm
    float* h2 = (float*)qkvb;        // overlay: qkv dead after edge scatter
    (void)ws_size; (void)n_in; (void)out_size;

    // 1) fp32 -> bf16 conversion of activations; weights packed to frag order
    k_cvt_bf16<<<(N * DIM + 255) / 256, 256, 0, stream>>>(h_one, hb, N * DIM);
    k_pack_w<<<(DIM * QKV_COLS + 255) / 256, 256, 0, stream>>>(W_qkv, wqkvp, QKV_COLS);
    k_pack_w<<<(DIM * DIM + 255) / 256, 256, 0, stream>>>(W_mlp, wmlpp, DIM);

    // 2) init reduction buffers (d_ws is poisoned; must re-init every call)
    k_fill<<<(N * HEADS + 255) / 256, 256, 0, stream>>>(segmax, -3.4e38f, N * HEADS);
    k_fill<<<(N * HEADS + 255) / 256, 256, 0, stream>>>(denom, 0.0f, N * HEADS);
    k_fill<<<(N * DIM + 255) / 256, 256, 0, stream>>>(attn, 0.0f, N * DIM);

    // 3) QKV projection (WMMA bf16): grid (N/64, 24/4)
    k_gemm_qkv<<<dim3(N / 64, QKV_COLS / 64), 128, 0, stream>>>(hb, wqkvp, b_qkv, qkvb);

    // 4) edge attention: scores+max, exp+sum, normalize+scatter
    int eblk = (EH + 255) / 256;
    k_edge_score<<<eblk, 256, 0, stream>>>(qkvb, e_i, e_j, A, segmax, EH);
    k_edge_exp<<<eblk, 256, 0, stream>>>(e_j, A, segmax, denom, EH);
    k_edge_scatter<<<eblk, 256, 0, stream>>>(qkvb, e_i, e_j, A, denom, attn, EH);

    // 5) h1 = LN1(h_one + attn)   (+ bf16 copy for MLP WMMA)
    k_add_ln<<<N / 4, 128, 0, stream>>>(h_one, attn, ln1_s, ln1_b, h1, h1b);

    // 6) h2 = h1 + silu(h1 @ W_mlp + b)   (WMMA bf16, fused epilogue)
    k_gemm_mlp<<<dim3(N / 64, DIM / 64), 128, 0, stream>>>(h1b, wmlpp, b_mlp, h1, h2);

    // 7) out = LN2(h2)
    k_add_ln<<<N / 4, 128, 0, stream>>>(h2, nullptr, ln2_s, ln2_b, out, nullptr);
}
